// ICMRecurrent_22016002359304
// MI455X (gfx1250) — compile-verified, matrix operand order checked by reading, not run on hardware
//
#include <hip/hip_runtime.h>
#include <hip/hip_bf16.h>
#include <math.h>

// ---------------------------------------------------------------------------
// Types for CDNA5 WMMA (wave32): A/B = v16h (16 x f16 = 8 VGPRs), C/D = v8f
// ---------------------------------------------------------------------------
typedef _Float16 v8h  __attribute__((ext_vector_type(8)));
typedef _Float16 v16h __attribute__((ext_vector_type(16)));
typedef float    v8f  __attribute__((ext_vector_type(8)));

#define DEVFN __device__ __forceinline__

// Async copy 16 bytes global -> LDS (per lane), tracked by ASYNCcnt.
DEVFN void async_g2l_b128(uint32_t lds_off, const void* gaddr) {
  asm volatile("global_load_async_to_lds_b128 %0, %1, off"
               :: "v"(lds_off), "v"(gaddr)
               : "memory");
}
DEVFN void wait_async0() {
  asm volatile("s_wait_asynccnt 0" ::: "memory");
}

// Load one 16x32 (MxK) f16 operand tile in CDNA5 WMMA A-layout from global.
// lane 0-15  : row = base + (lane&15), halves 0..7 = K kk+0..7,  8..15 = kk+16..23
// lane 16-31 : row = base + (lane&15), halves 0..7 = K kk+8..15, 8..15 = kk+24..31
DEVFN v16h load_tile16x32(const _Float16* __restrict__ base, int row, int ldk,
                          int kk, int lane) {
  const _Float16* p = base + (size_t)row * ldk + kk + ((lane >> 4) << 3);
  v8h lo = *(const v8h*)(p);
  v8h hi = *(const v8h*)(p + 16);
  v16h r;
#pragma unroll
  for (int i = 0; i < 8; ++i) { r[i] = lo[i]; r[8 + i] = hi[i]; }
  return r;
}

DEVFN float sigm(float x) { return 1.0f / (1.0f + expf(-x)); }

template <int ACT>
DEVFN float apply_act(float x) {
  if (ACT == 1) return fmaxf(x, 0.0f);                       // relu
  if (ACT == 2) return tanhf(x);                             // tanh
  if (ACT == 3) return (x > 20.0f) ? x : log1pf(expf(x));    // softplus
  return x;
}

// ---------------------------------------------------------------------------
// Generic NT GEMM:  out[m,n] = act( sum_k X[m,k]*W[n,k] + bias[n] )
// Block = 8 waves, block tile 128(M) x 32(N). The 32x32 W tile is shared by
// all 8 waves: double-buffered in LDS, filled with async global->LDS b128
// copies (ASYNCcnt), consumed with per-lane ds_load_b128 gathers.
// Each wave runs two v_wmma_f32_16x16x32_f16 per K step.
// M % 128 == 0, N % 32 == 0, K % 32 == 0.  blockDim = 256.
// ---------------------------------------------------------------------------
template <int ACT>
__global__ void gemm_nt_wmma(const _Float16* __restrict__ X,
                             const _Float16* __restrict__ W,
                             const float* __restrict__ bias,
                             _Float16* __restrict__ outH,
                             float* __restrict__ outF,
                             int M, int N, int K) {
  __shared__ _Float16 sB[2][32 * 32];  // [buf][row*32 + k], 2 KB each

  const int tid  = threadIdx.x;
  const int lane = tid & 31;
  const int wv   = tid >> 5;                 // 0..7
  const int nblk = N >> 5;
  const int bm   = blockIdx.x / nblk;        // 128-row block index
  const int n0   = (blockIdx.x % nblk) << 5;
  const int m0   = (bm << 7) + (wv << 4);    // this wave's 16 rows

  const int arow = m0 + (lane & 15);

  // async-stage first W tile (128 lanes x 16B = 2 KB)
  const int srow = tid >> 2;                 // 0..31
  const int skp  = (tid & 3) << 3;           // 0,8,16,24 halves
  if (tid < 128)
    async_g2l_b128((uint32_t)(uintptr_t)&sB[0][srow * 32 + skp],
                   W + (size_t)(n0 + srow) * K + skp);

  v8f acc0 = {};
  v8f acc1 = {};
  const int nk = K >> 5;
  for (int i = 0; i < nk; ++i) {
    if (tid < 128) wait_async0();            // our async fill of buf i&1 done
    __syncthreads();
    if (i + 1 < nk && tid < 128)             // prefetch next tile into buf^1
      async_g2l_b128((uint32_t)(uintptr_t)&sB[(i + 1) & 1][srow * 32 + skp],
                     W + (size_t)(n0 + srow) * K + ((i + 1) << 5) + skp);

    const int kk = i << 5;
    v16h a = load_tile16x32(X, arow, K, kk, lane);

    // per-lane WMMA B-operand gathers from LDS (64 B row stride)
    const _Float16* bp0 =
        &sB[i & 1][(lane & 15) * 32 + ((lane >> 4) << 3)];
    const _Float16* bp1 = bp0 + 16 * 32;
    v16h b0, b1;
    {
      v8h lo = *(const v8h*)(bp0);
      v8h hi = *(const v8h*)(bp0 + 16);
#pragma unroll
      for (int q = 0; q < 8; ++q) { b0[q] = lo[q]; b0[8 + q] = hi[q]; }
    }
    {
      v8h lo = *(const v8h*)(bp1);
      v8h hi = *(const v8h*)(bp1 + 16);
#pragma unroll
      for (int q = 0; q < 8; ++q) { b1[q] = lo[q]; b1[8 + q] = hi[q]; }
    }
    acc0 = __builtin_amdgcn_wmma_f32_16x16x32_f16(false, a, false, b0,
                                                  (short)0, acc0, false, false);
    acc1 = __builtin_amdgcn_wmma_f32_16x16x32_f16(false, a, false, b1,
                                                  (short)0, acc1, false, false);
    __syncthreads();                         // reads done before buf reuse
  }

  const int col0  = n0 + (lane & 15);
  const int col1  = col0 + 16;
  const float bi0 = bias ? bias[col0] : 0.0f;
  const float bi1 = bias ? bias[col1] : 0.0f;
  const int rbase = m0 + ((lane >> 4) << 3);
#pragma unroll
  for (int r = 0; r < 8; ++r) {
    const int row = rbase + r;
    float v0 = apply_act<ACT>(acc0[r] + bi0);
    float v1 = apply_act<ACT>(acc1[r] + bi1);
    if (outF) {
      outF[(size_t)row * N + col0] = v0;
      outF[(size_t)row * N + col1] = v1;
    }
    if (outH) {
      outH[(size_t)row * N + col0] = (_Float16)v0;
      outH[(size_t)row * N + col1] = (_Float16)v1;
    }
  }
}

// ---------------------------------------------------------------------------
// Fused LSTM step: gates = gx[t] + (mask*h) @ Whh^T ; cell update; record
// hidden_outputs[t] = (masked h, masked c). One wave owns a 16x16 (b,f)
// block and computes all 4 gate columns for it (j, j+256, j+512, j+768).
// grid = 32 blocks x 128 threads = 128 waves  (8 m-tiles x 16 j-tiles).
// ---------------------------------------------------------------------------
__global__ void lstm_step_kernel(const _Float16* __restrict__ Whh_h, // (1024,256) f16
                                 const float* __restrict__ gx,       // (1025,128,1024)
                                 const int* __restrict__ dones,      // (1025,128)
                                 _Float16* __restrict__ hbuf,        // (1026,128,256) f16
                                 float* __restrict__ cbuf,           // (128,256) f32
                                 float* __restrict__ hid_out,        // (1025,2,128,256)
                                 int t) {
  const int lane = threadIdx.x & 31;
  const int wid  = blockIdx.x * (blockDim.x >> 5) + (threadIdx.x >> 5); // 0..127
  const int m0 = (wid >> 4) << 4;   // batch tile
  const int j0 = (wid & 15) << 4;   // feature tile

  const _Float16* hprev = hbuf + (size_t)t * (128 * 256);
  const int* dn = dones + (size_t)t * 128;

  const int am = m0 + (lane & 15);
  const _Float16 maskA = dn[am] ? (_Float16)0.0f : (_Float16)1.0f;

  v8f acc[4] = {};
  for (int kk = 0; kk < 256; kk += 32) {
    v16h a = load_tile16x32(hprev, am, 256, kk, lane);
#pragma unroll
    for (int i = 0; i < 16; ++i) a[i] *= maskA;
#pragma unroll
    for (int g = 0; g < 4; ++g) {
      const int brow = g * 256 + j0 + (lane & 15);
      v16h b = load_tile16x32(Whh_h, brow, 256, kk, lane);
      acc[g] = __builtin_amdgcn_wmma_f32_16x16x32_f16(false, a, false, b,
                                                      (short)0, acc[g], false, false);
    }
  }

  const int col   = j0 + (lane & 15);
  const int rbase = m0 + ((lane >> 4) << 3);
  float* hidh = hid_out + (size_t)t * (2 * 128 * 256);
  float* hidc = hidh + 128 * 256;

  float cm[8];
#pragma unroll
  for (int r = 0; r < 8; ++r) {
    const int b = rbase + r;
    const float mr = dn[b] ? 0.0f : 1.0f;
    const float hm = (float)hprev[b * 256 + col] * mr;
    const float c  = cbuf[b * 256 + col] * mr;
    hidh[b * 256 + col] = hm;
    hidc[b * 256 + col] = c;
    cm[r] = c;
  }

  const float* gxt = gx + (size_t)t * (128 * 1024);
  _Float16* hnext = hbuf + (size_t)(t + 1) * (128 * 256);
#pragma unroll
  for (int r = 0; r < 8; ++r) {
    const int b = rbase + r;
    const float* gb = gxt + b * 1024 + col;
    const float gi = acc[0][r] + gb[0];
    const float gf = acc[1][r] + gb[256];
    const float gg = acc[2][r] + gb[512];
    const float go = acc[3][r] + gb[768];
    const float cn = sigm(gf) * cm[r] + sigm(gi) * tanhf(gg);
    const float hn = sigm(go) * tanhf(cn);
    cbuf[b * 256 + col]  = cn;
    hnext[b * 256 + col] = (_Float16)hn;
  }
}

// ---------------------------------------------------------------------------
// Small helper kernels
// ---------------------------------------------------------------------------
__global__ void cvt_f32_f16(const float* __restrict__ src, _Float16* __restrict__ dst,
                            size_t n) {
  for (size_t i = blockIdx.x * (size_t)blockDim.x + threadIdx.x; i < n;
       i += (size_t)gridDim.x * blockDim.x)
    dst[i] = (_Float16)src[i];
}

__global__ void pad_w_f16(const float* __restrict__ src, _Float16* __restrict__ dst,
                          int src_rows, int dst_rows, int src_cols, int dst_cols) {
  size_t n = (size_t)dst_rows * dst_cols;
  for (size_t i = blockIdx.x * (size_t)blockDim.x + threadIdx.x; i < n;
       i += (size_t)gridDim.x * blockDim.x) {
    int r = (int)(i / dst_cols), c = (int)(i % dst_cols);
    dst[i] = (r < src_rows && c < src_cols) ? (_Float16)src[(size_t)r * src_cols + c]
                                            : (_Float16)0.0f;
  }
}

__global__ void pad_bias_f32(const float* __restrict__ src, float* __restrict__ dst,
                             int src_n, int dst_n) {
  int i = blockIdx.x * blockDim.x + threadIdx.x;
  if (i < dst_n) dst[i] = (i < src_n) ? src[i] : 0.0f;
}

__global__ void combine_bias(const float* __restrict__ a, const float* __restrict__ b,
                             float* __restrict__ dst, int n) {
  int i = blockIdx.x * blockDim.x + threadIdx.x;
  if (i < n) dst[i] = a[i] + b[i];
}

__global__ void zero_f16(_Float16* p, size_t n) {
  for (size_t i = blockIdx.x * (size_t)blockDim.x + threadIdx.x; i < n;
       i += (size_t)gridDim.x * blockDim.x)
    p[i] = (_Float16)0.0f;
}
__global__ void zero_f32(float* p, size_t n) {
  for (size_t i = blockIdx.x * (size_t)blockDim.x + threadIdx.x; i < n;
       i += (size_t)gridDim.x * blockDim.x)
    p[i] = 0.0f;
}
__global__ void init_scalars(float* out) {
  if (threadIdx.x == 0) { out[0] = 0.0f; out[1] = 0.0f; }
}

// xfm[row, 0:256]=sf, [256:264]=loc, [264:272]=scale, [272:288]=0
// sf[row] = lstm_out[t] = hbuf slot t+1  -> hbuf[(row+128)*256 + col]
__global__ void build_xfm_kernel(const _Float16* __restrict__ hbuf,
                                 const float* __restrict__ loc,
                                 const float* __restrict__ scale,
                                 _Float16* __restrict__ xfm) {
  const int row = blockIdx.x;   // 0..131071
  const int col = threadIdx.x;  // blockDim = 288
  _Float16 v;
  if (col < 256)      v = hbuf[((size_t)row + 128) * 256 + col];
  else if (col < 264) v = (_Float16)loc[(size_t)row * 8 + (col - 256)];
  else if (col < 272) v = (_Float16)scale[(size_t)row * 8 + (col - 264)];
  else                v = (_Float16)0.0f;
  xfm[(size_t)row * 288 + col] = v;
}

__global__ void build_xi_kernel(const _Float16* __restrict__ hbuf,
                                const _Float16* __restrict__ pnH,
                                _Float16* __restrict__ xi) {
  const int row = blockIdx.x;   // 0..131071
  const int col = threadIdx.x;  // blockDim = 512
  _Float16 v = (col < 256) ? hbuf[((size_t)row + 128) * 256 + col]
                           : pnH[(size_t)row * 256 + (col - 256)];
  xi[(size_t)row * 512 + col] = v;
}

// forward loss + intrinsic reward.  nsf[row] = lstm_out[t+1] = hbuf slot t+2
__global__ void fwd_loss_kernel(const float* __restrict__ pn,
                                const _Float16* __restrict__ hbuf,
                                float* __restrict__ out) {
  __shared__ float red[256];
  const int row = blockIdx.x;   // 0..131071
  const int k   = threadIdx.x;  // 0..255
  const float p   = pn[(size_t)row * 256 + k];
  const float nsf = (float)hbuf[((size_t)row + 256) * 256 + k];
  const float d   = p - nsf;
  red[k] = d * d;
  __syncthreads();
  for (int s = 128; s > 0; s >>= 1) {
    if (k < s) red[k] += red[k + s];
    __syncthreads();
  }
  if (k == 0) {
    const float sum = red[0];
    out[2 + row] = 0.5f * (sum * (1.0f / 256.0f));                       // N_SCALE
    atomicAdd(&out[0], 0.2f * sum * (1.0f / (131072.0f * 256.0f)));      // BETA
  }
}

__global__ void inv_loss_kernel(const float* __restrict__ ploc,
                                const float* __restrict__ pscale,
                                const float* __restrict__ loc,
                                const float* __restrict__ scale,
                                float* __restrict__ out) {
  __shared__ float red[256];
  float acc = 0.0f;
  const int total = 1024 * 128 * 8;
  for (int idx = blockIdx.x * blockDim.x + threadIdx.x; idx < total;
       idx += gridDim.x * blockDim.x) {
    const int row = idx >> 3, a = idx & 7;
    const float dl = ploc[(size_t)row * 32 + a] - loc[idx];
    const float ds = pscale[(size_t)row * 32 + a] - scale[idx];
    acc += dl * dl + ds * ds;
  }
  red[threadIdx.x] = acc;
  __syncthreads();
  for (int s = 128; s > 0; s >>= 1) {
    if (threadIdx.x < s) red[threadIdx.x] += red[threadIdx.x + s];
    __syncthreads();
  }
  if (threadIdx.x == 0)
    atomicAdd(&out[1], 0.8f * red[0] * (1.0f / 1048576.0f));  // (1-BETA)
}

// ---------------------------------------------------------------------------
// Host: orchestration
// ---------------------------------------------------------------------------
extern "C" void kernel_launch(void* const* d_in, const int* in_sizes, int n_in,
                              void* d_out, int out_size, void* d_ws, size_t ws_size,
                              hipStream_t stream) {
  (void)in_sizes; (void)n_in; (void)out_size; (void)ws_size;

  const float* states = (const float*)d_in[0];   // (1025,128,128)
  const float* loc    = (const float*)d_in[1];   // (1024,128,8)
  const float* scale  = (const float*)d_in[2];   // (1024,128,8)
  const int*   dones  = (const int*)d_in[3];     // (1025,128)
  const float* fe_W1  = (const float*)d_in[4];   // (256,128)
  const float* fe_b1  = (const float*)d_in[5];
  const float* fe_W2  = (const float*)d_in[6];   // (256,256)
  const float* fe_b2  = (const float*)d_in[7];
  const float* Wih    = (const float*)d_in[8];   // (1024,256)
  const float* bih    = (const float*)d_in[9];
  const float* Whh    = (const float*)d_in[10];  // (1024,256)
  const float* bhh    = (const float*)d_in[11];
  const float* fm_W1  = (const float*)d_in[12];  // (256,272)
  const float* fm_b1  = (const float*)d_in[13];
  const float* fm_W2  = (const float*)d_in[14];  // (256,256)
  const float* fm_b2  = (const float*)d_in[15];
  const float* im_W1  = (const float*)d_in[16];  // (256,512)
  const float* im_b1  = (const float*)d_in[17];
  const float* im_W2  = (const float*)d_in[18];  // (8,256)
  const float* im_b2  = (const float*)d_in[19];
  const float* is_W1  = (const float*)d_in[20];  // (256,512)
  const float* is_b1  = (const float*)d_in[21];
  const float* is_W2  = (const float*)d_in[22];  // (8,256)
  const float* is_b2  = (const float*)d_in[23];

  float* out = (float*)d_out;
  // d_out layout: [0]=forward_loss [1]=inverse_loss [2..2+131072)=intrinsic
  //               then hidden_outputs (1025,2,128,256)
  float* hid_out = out + 2 + 131072;

  const int SB  = 1025 * 128;   // 131200
  const int SBm = 1024 * 128;   // 131072

  char* wp = (char*)d_ws;
  auto carve = [&](size_t bytes) -> char* {
    char* r = wp;
    wp += (bytes + 255) & ~(size_t)255;
    return r;
  };
  _Float16* fe_W1h = (_Float16*)carve((size_t)256 * 128 * 2);
  _Float16* fe_W2h = (_Float16*)carve((size_t)256 * 256 * 2);
  _Float16* Wih_h  = (_Float16*)carve((size_t)1024 * 256 * 2);
  _Float16* Whh_h  = (_Float16*)carve((size_t)1024 * 256 * 2);
  _Float16* fm_W1p = (_Float16*)carve((size_t)256 * 288 * 2);
  _Float16* fm_W2h = (_Float16*)carve((size_t)256 * 256 * 2);
  _Float16* im_W1h = (_Float16*)carve((size_t)256 * 512 * 2);
  _Float16* im_W2p = (_Float16*)carve((size_t)32 * 256 * 2);
  _Float16* is_W1h = (_Float16*)carve((size_t)256 * 512 * 2);
  _Float16* is_W2p = (_Float16*)carve((size_t)32 * 256 * 2);
  float*    bias_g = (float*)carve((size_t)1024 * 4);      // bih + bhh
  float*    im_b2p = (float*)carve((size_t)32 * 4);
  float*    is_b2p = (float*)carve((size_t)32 * 4);
  _Float16* st_h   = (_Float16*)carve((size_t)SB * 128 * 2);
  _Float16* f1_h   = (_Float16*)carve((size_t)SB * 256 * 2);
  _Float16* f_h    = (_Float16*)carve((size_t)SB * 256 * 2);
  float*    gx     = (float*)carve((size_t)SB * 1024 * 4); // gates_x + biases
  _Float16* hbuf   = (_Float16*)carve((size_t)1026 * 128 * 256 * 2); // slot0 = h0
  float*    cbuf   = (float*)carve((size_t)128 * 256 * 4);
  _Float16* xfm    = (_Float16*)carve((size_t)SBm * 288 * 2);
  _Float16* fmhid  = (_Float16*)carve((size_t)SBm * 256 * 2);
  float*    pnF    = (float*)carve((size_t)SBm * 256 * 4);
  _Float16* pnH    = (_Float16*)carve((size_t)SBm * 256 * 2);
  _Float16* xi     = (_Float16*)carve((size_t)SBm * 512 * 2);
  _Float16* imhid  = (_Float16*)carve((size_t)SBm * 256 * 2);
  _Float16* ishid  = (_Float16*)carve((size_t)SBm * 256 * 2);
  float*    ploc   = (float*)carve((size_t)SBm * 32 * 4);
  float*    pscale = (float*)carve((size_t)SBm * 32 * 4);

  auto gemm = [&](int act, const _Float16* X, const _Float16* W, const float* bias,
                  _Float16* oh, float* of, int M, int N, int K) {
    const int blocks = (M / 128) * (N / 32);
    switch (act) {
      case 0: gemm_nt_wmma<0><<<blocks, 256, 0, stream>>>(X, W, bias, oh, of, M, N, K); break;
      case 1: gemm_nt_wmma<1><<<blocks, 256, 0, stream>>>(X, W, bias, oh, of, M, N, K); break;
      case 2: gemm_nt_wmma<2><<<blocks, 256, 0, stream>>>(X, W, bias, oh, of, M, N, K); break;
      case 3: gemm_nt_wmma<3><<<blocks, 256, 0, stream>>>(X, W, bias, oh, of, M, N, K); break;
    }
  };

  // ---- prep: convert / pad weights, combine biases, zero state ----
  init_scalars<<<1, 64, 0, stream>>>(out);
  cvt_f32_f16<<<128, 256, 0, stream>>>(fe_W1, fe_W1h, (size_t)256 * 128);
  cvt_f32_f16<<<256, 256, 0, stream>>>(fe_W2, fe_W2h, (size_t)256 * 256);
  cvt_f32_f16<<<1024, 256, 0, stream>>>(Wih, Wih_h, (size_t)1024 * 256);
  cvt_f32_f16<<<1024, 256, 0, stream>>>(Whh, Whh_h, (size_t)1024 * 256);
  cvt_f32_f16<<<256, 256, 0, stream>>>(fm_W2, fm_W2h, (size_t)256 * 256);
  cvt_f32_f16<<<512, 256, 0, stream>>>(im_W1, im_W1h, (size_t)256 * 512);
  cvt_f32_f16<<<512, 256, 0, stream>>>(is_W1, is_W1h, (size_t)256 * 512);
  pad_w_f16<<<288, 256, 0, stream>>>(fm_W1, fm_W1p, 256, 256, 272, 288);
  pad_w_f16<<<32, 256, 0, stream>>>(im_W2, im_W2p, 8, 32, 256, 256);
  pad_w_f16<<<32, 256, 0, stream>>>(is_W2, is_W2p, 8, 32, 256, 256);
  combine_bias<<<4, 256, 0, stream>>>(bih, bhh, bias_g, 1024);
  pad_bias_f32<<<1, 64, 0, stream>>>(im_b2, im_b2p, 8, 32);
  pad_bias_f32<<<1, 64, 0, stream>>>(is_b2, is_b2p, 8, 32);
  cvt_f32_f16<<<8192, 256, 0, stream>>>(states, st_h, (size_t)SB * 128);
  zero_f16<<<128, 256, 0, stream>>>(hbuf, (size_t)128 * 256);  // h0 = 0
  zero_f32<<<128, 256, 0, stream>>>(cbuf, (size_t)128 * 256);  // c0 = 0

  // ---- feature extractor (parallel GEMMs) ----
  gemm(1, st_h, fe_W1h, fe_b1, f1_h, nullptr, SB, 256, 128);   // relu
  gemm(1, f1_h, fe_W2h, fe_b2, f_h, nullptr, SB, 256, 256);    // relu
  // gates_x = f @ Wih^T + (bih + bhh)  (parallel part of the LSTM)
  gemm(0, f_h, Wih_h, bias_g, nullptr, gx, SB, 1024, 256);

  // ---- sequential LSTM: 1025 fused steps ----
  for (int t = 0; t < 1025; ++t)
    lstm_step_kernel<<<32, 128, 0, stream>>>(Whh_h, gx, dones, hbuf, cbuf,
                                             hid_out, t);

  // ---- forward model ----
  build_xfm_kernel<<<SBm, 288, 0, stream>>>(hbuf, loc, scale, xfm);
  gemm(1, xfm, fm_W1p, fm_b1, fmhid, nullptr, SBm, 256, 288);  // relu
  gemm(0, fmhid, fm_W2h, fm_b2, pnH, pnF, SBm, 256, 256);      // pred_next

  // ---- inverse models ----
  build_xi_kernel<<<SBm, 512, 0, stream>>>(hbuf, pnH, xi);
  gemm(1, xi, im_W1h, im_b1, imhid, nullptr, SBm, 256, 512);   // relu
  gemm(2, imhid, im_W2p, im_b2p, nullptr, ploc, SBm, 32, 256); // tanh
  gemm(1, xi, is_W1h, is_b1, ishid, nullptr, SBm, 256, 512);   // relu
  gemm(3, ishid, is_W2p, is_b2p, nullptr, pscale, SBm, 32, 256); // softplus

  // ---- losses + intrinsic reward ----
  fwd_loss_kernel<<<SBm, 256, 0, stream>>>(pnF, hbuf, out);
  inv_loss_kernel<<<4096, 256, 0, stream>>>(ploc, pscale, loc, scale, out);
}
